// Zero2Zero_80736795230596
// MI455X (gfx1250) — compile-verified, matrix operand order checked by reading, not run on hardware
//
#include <hip/hip_runtime.h>

typedef __attribute__((ext_vector_type(16))) __bf16 v16bf;
typedef __attribute__((ext_vector_type(8)))  __bf16 v8bf;
typedef __attribute__((ext_vector_type(8)))  float  v8f;

#define EDGES_TOTAL (384*384)
#define TILE_M 128
#define SA 136   /* bf16 elems per row (128 + 8 pad), 272B = 16B aligned   */
#define SF 132   /* f32 elems per row (128 + 4 pad), 528B = 16B aligned    */
#define SX 34    /* f32 elems per row for x tile (32 + 2 pad)              */
#define SO 34    /* f32 elems per row for out staging                      */

__device__ __forceinline__ unsigned short f2bf_bits(float f) {
    unsigned u = __float_as_uint(f);
    u += 0x7FFFu + ((u >> 16) & 1u);   // round-to-nearest-even to bf16
    return (unsigned short)(u >> 16);
}

// Build a 16x32 (or 32x16) bf16 WMMA fragment: per-lane two contiguous
// 8-element (16B) chunks at K = kb..kb+7 and K = kb+16..kb+23 (ISA layout:
// lane-half h covers K = h*8+{0..7} and 16+h*8+{0..7}).
__device__ __forceinline__ v16bf ldfrag(const unsigned short* p0, const unsigned short* p1) {
    v8bf a = *(const v8bf*)p0;
    v8bf b = *(const v8bf*)p1;
    v16bf r;
#pragma unroll
    for (int i = 0; i < 8; ++i) { r[i] = a[i]; r[i + 8] = b[i]; }
    return r;
}

// ---- Weight prep: transpose to (N,K) and convert fp32 -> bf16 ----
__global__ void prep_weights_kernel(const float* __restrict__ W1,
                                    const float* __restrict__ W2,
                                    const float* __restrict__ W3,
                                    unsigned short* __restrict__ W1t,
                                    unsigned short* __restrict__ W2t,
                                    unsigned short* __restrict__ W3t) {
    int t = blockIdx.x * blockDim.x + threadIdx.x;
    if (t < 128 * 128) {
        int n = t >> 7, k = t & 127;
        W1t[n * 128 + k] = f2bf_bits(W1[k * 128 + n]);   // W1: (K=128, N=128)
        W2t[n * 128 + k] = f2bf_bits(W2[k * 128 + n]);   // W2: (K=128, N=128)
    }
    if (t < 1024 * 128) {
        int n = t >> 7, k = t & 127;
        W3t[n * 128 + k] = f2bf_bits(W3[k * 1024 + n]);  // W3: (K=128, N=1024)
    }
}

// ---- Main fused kernel: one 128-edge tile per workgroup, 8 waves ----
__global__ void __launch_bounds__(256, 1)
edge_mlp_kernel(const float* __restrict__ eg, const float* __restrict__ x,
                const float* __restrict__ b1, const float* __restrict__ g1, const float* __restrict__ be1,
                const float* __restrict__ b2, const float* __restrict__ g2, const float* __restrict__ be2,
                const float* __restrict__ b3,
                const unsigned short* __restrict__ W1t,
                const unsigned short* __restrict__ W2t,
                const unsigned short* __restrict__ W3t,
                float* __restrict__ out) {
    __shared__ __align__(16) unsigned char smem[34816 + 17408 + 67584];
    unsigned short* sA = (unsigned short*)smem;          // [128][SA] bf16 activations
    float* sX = (float*)(smem + 34816);                  // [128][SX] x tile
    float* sF = (float*)(smem + 34816 + 17408);          // [128][SF] fp32 GEMM results
    float* sO = sF;                                      // [128][SO] out staging (reuse)

    const int tid  = threadIdx.x;
    const int wave = tid >> 5;
    const int lane = tid & 31;
    const int l16  = lane & 15;
    const int hh   = lane >> 4;
    const long tile = (long)blockIdx.x * TILE_M;

    // Load edge tile (128x128 fp32, fully coalesced) -> bf16 LDS; load x tile.
    {
        const float* src = eg + tile * 128;
#pragma unroll
        for (int it = 0; it < 64; ++it) {
            int g = tid + it * 256;
            sA[(g >> 7) * SA + (g & 127)] = f2bf_bits(src[g]);
        }
        const float* xs = x + tile * 32;
#pragma unroll
        for (int it = 0; it < 16; ++it) {
            int g = tid + it * 256;
            sX[(g >> 5) * SX + (g & 31)] = xs[g];
        }
    }
    __syncthreads();

    // ---- Stages 1 & 2: H = GELU(LN(A @ W^T + b)) ----
    for (int stage = 0; stage < 2; ++stage) {
        const unsigned short* Wt  = stage ? W2t : W1t;
        const float* bias = stage ? b2  : b1;
        const float* gam  = stage ? g2  : g1;
        const float* bet  = stage ? be2 : be1;

        // A-fragments for this wave's 16-row band (cached over all n-tiles)
        v16bf afr[4];
        const unsigned short* arow = sA + (wave * 16 + l16) * SA;
#pragma unroll
        for (int ks = 0; ks < 4; ++ks) {
            int kb = ks * 32 + hh * 8;
            afr[ks] = ldfrag(arow + kb, arow + kb + 16);
        }
#pragma unroll
        for (int nt = 0; nt < 8; ++nt) {
            int ncol = nt * 16 + l16;
            float bv = bias[ncol];
            v8f c;
#pragma unroll
            for (int i = 0; i < 8; ++i) c[i] = bv;     // bias folded into C
            const unsigned short* brow = Wt + ncol * 128;
#pragma unroll
            for (int ks = 0; ks < 4; ++ks) {
                int kb = ks * 32 + hh * 8;
                v16bf bfr = ldfrag(brow + kb, brow + kb + 16);
                c = __builtin_amdgcn_wmma_f32_16x16x32_bf16(
                        false, afr[ks], false, bfr, (short)0, c, false, false);
            }
#pragma unroll
            for (int r = 0; r < 8; ++r)                // C layout: VGPR r -> M = r (+8 for hi half)
                sF[(wave * 16 + hh * 8 + r) * SF + ncol] = c[r];
        }
        __syncthreads();

        // LayerNorm (population var) + exact GELU, fp32 -> bf16 into sA.
        {
            int row  = tid >> 1;
            int half = tid & 1;                        // adjacent lanes pair via shfl_xor(1)
            const float* fr = sF + row * SF + half * 64;
            float s = 0.f, ss = 0.f;
#pragma unroll
            for (int i = 0; i < 64; ++i) { float v = fr[i]; s += v; ss += v * v; }
            s  += __shfl_xor(s, 1, 32);
            ss += __shfl_xor(ss, 1, 32);
            float mean = s * (1.f / 128.f);
            float var  = ss * (1.f / 128.f) - mean * mean;
            float rstd = rsqrtf(var + 1e-5f);
            unsigned short* dr = sA + row * SA + half * 64;
#pragma unroll
            for (int i = 0; i < 64; ++i) {
                int cix = half * 64 + i;
                float v = (fr[i] - mean) * rstd * gam[cix] + bet[cix];
                float g = 0.5f * v * (1.f + erff(v * 0.70710678118654752f));
                dr[i] = f2bf_bits(g);
            }
        }
        __syncthreads();
    }

    // ---- Stage 3: per-o slice of k = H @ W3^T + b3, contracted with x ----
    v16bf afr[4];
    const unsigned short* arow = sA + (wave * 16 + l16) * SA;
#pragma unroll
    for (int ks = 0; ks < 4; ++ks) {
        int kb = ks * 32 + hh * 8;
        afr[ks] = ldfrag(arow + kb, arow + kb + 16);
    }

    for (int o = 0; o < 32; ++o) {
        float bva = b3[o * 32 + l16];
        float bvb = b3[o * 32 + 16 + l16];
        v8f c0, c1;
#pragma unroll
        for (int i = 0; i < 8; ++i) { c0[i] = bva; c1[i] = bvb; }
        const unsigned short* br0 = W3t + (o * 32 + l16) * 128;        // i = 0..15
        const unsigned short* br1 = W3t + (o * 32 + 16 + l16) * 128;   // i = 16..31
#pragma unroll
        for (int ks = 0; ks < 4; ++ks) {
            int kb = ks * 32 + hh * 8;
            v16bf bf0 = ldfrag(br0 + kb, br0 + kb + 16);
            c0 = __builtin_amdgcn_wmma_f32_16x16x32_bf16(
                    false, afr[ks], false, bf0, (short)0, c0, false, false);
        }
#pragma unroll
        for (int ks = 0; ks < 4; ++ks) {
            int kb = ks * 32 + hh * 8;
            v16bf bf1 = ldfrag(br1 + kb, br1 + kb + 16);
            c1 = __builtin_amdgcn_wmma_f32_16x16x32_bf16(
                    false, afr[ks], false, bf1, (short)0, c1, false, false);
        }
        // out[e][o] = sum_i k[e][o*32+i] * x[e][i]; lane l16 holds i=l16 (c0) and i=l16+16 (c1)
#pragma unroll
        for (int r = 0; r < 8; ++r) {
            int e = wave * 16 + hh * 8 + r;
            const float* xr = sX + e * SX;
            float acc = c0[r] * xr[l16] + c1[r] * xr[l16 + 16];
            acc += __shfl_xor(acc, 1, 32);
            acc += __shfl_xor(acc, 2, 32);
            acc += __shfl_xor(acc, 4, 32);
            acc += __shfl_xor(acc, 8, 32);
            if (l16 == 0) sO[e * SO + o] = acc;
        }
    }
    __syncthreads();

    // Coalesced store of the 128x32 output tile.
    float* dst = out + tile * 32;
#pragma unroll
    for (int it = 0; it < 16; ++it) {
        int g = tid + it * 256;
        dst[g] = sO[(g >> 5) * SO + (g & 31)];
    }
}

extern "C" void kernel_launch(void* const* d_in, const int* in_sizes, int n_in,
                              void* d_out, int out_size, void* d_ws, size_t ws_size,
                              hipStream_t stream) {
    const float* x   = (const float*)d_in[0];
    const float* eg  = (const float*)d_in[1];
    const float* W1  = (const float*)d_in[2];
    const float* b1  = (const float*)d_in[3];
    const float* g1  = (const float*)d_in[4];
    const float* be1 = (const float*)d_in[5];
    const float* W2  = (const float*)d_in[6];
    const float* b2  = (const float*)d_in[7];
    const float* g2  = (const float*)d_in[8];
    const float* be2 = (const float*)d_in[9];
    const float* W3  = (const float*)d_in[10];
    const float* b3  = (const float*)d_in[11];

    unsigned short* W1t = (unsigned short*)d_ws;         //  32 KB
    unsigned short* W2t = W1t + 128 * 128;               //  32 KB
    unsigned short* W3t = W2t + 128 * 128;               // 256 KB

    prep_weights_kernel<<<(1024 * 128) / 256, 256, 0, stream>>>(W1, W2, W3, W1t, W2t, W3t);
    edge_mlp_kernel<<<EDGES_TOTAL / TILE_M, 256, 0, stream>>>(
        eg, x, b1, g1, be1, b2, g2, be2, b3, W1t, W2t, W3t, (float*)d_out);
}
